// HGAT_71554155151660
// MI455X (gfx1250) — matrix operand order repaired and checked by reading, not compiled, not run on hardware
//
#include <hip/hip_runtime.h>
#include <math.h>

#define NW 100000
#define NT 20000
#define ND 5000
#define E_WW 1600000
#define E_BU 1000000
#define E_SE 1000000
#define E_CR 20000

#define NWPAD 100096   // cdiv(NW,128)*128
#define NDPAD 5120     // cdiv(ND,128)*128

typedef __attribute__((ext_vector_type(16))) _Float16 v16h;
typedef __attribute__((ext_vector_type(8)))  _Float16 v8h;
typedef __attribute__((ext_vector_type(8)))  float    v8f;

union AB16 { v16h v; v8h h[2]; };

// ---------------------------------------------------------------------------
// Pack X[N,K] f32 -> Xh[Npad,Kpad] f16, zero padded. (Kpad mult of 32)
// ---------------------------------------------------------------------------
__global__ void cvt_rows(const float* __restrict__ X, _Float16* __restrict__ Xh,
                         int N, int K, int Kpad, int Npad) {
  int idx = blockIdx.x * blockDim.x + threadIdx.x;
  if (idx >= Npad * Kpad) return;
  int row = idx / Kpad, k = idx % Kpad;
  float v = (row < N && k < K) ? X[(size_t)row * K + k] : 0.0f;
  Xh[idx] = (_Float16)v;
}

// Pack W[K,64] f32 -> Wt[64,Kpad] f16 transposed, zero padded.
__global__ void cvt_wt(const float* __restrict__ W, _Float16* __restrict__ Wt,
                       int K, int Kpad) {
  int idx = blockIdx.x * blockDim.x + threadIdx.x;
  if (idx >= 64 * Kpad) return;
  int n = idx / Kpad, k = idx % Kpad;
  Wt[idx] = (_Float16)((k < K) ? W[(size_t)k * 64 + n] : 0.0f);
}

// ---------------------------------------------------------------------------
// WMMA GEMM: Y[N,64] = Xh[Npad,KPAD] @ Wt[64,KPAD]^T ; f16 wmma, f32 accum.
// One wave per 16-row tile; 4 accumulators cover the 64 output columns.
// KPAD is compile-time so the k-loop fully unrolls; per k-step all 10 b128
// loads (A:2, B:4x2) are issued before the 4 wmma ops to allow partial
// loadcnt waits / load-wmma overlap.
// ---------------------------------------------------------------------------
template <int KPAD>
__global__ __launch_bounds__(256) void gemm_h64(const _Float16* __restrict__ Xh,
                                                const _Float16* __restrict__ Wt,
                                                float* __restrict__ Y,
                                                int N) {
  const int wave = threadIdx.x >> 5;
  const int lane = threadIdx.x & 31;
  const int m    = lane & 15;
  const int half = lane >> 4;
  const int row0 = (blockIdx.x * 8 + wave) * 16;

  const _Float16* xr = Xh + (size_t)(row0 + m) * KPAD;

  v8f c[4] = {v8f{}, v8f{}, v8f{}, v8f{}};

#pragma unroll
  for (int k0 = 0; k0 < KPAD; k0 += 32) {
    AB16 a, b[4];
    a.h[0] = *(const v8h*)(xr + k0 + half * 8);
    a.h[1] = *(const v8h*)(xr + k0 + 16 + half * 8);
#pragma unroll
    for (int t = 0; t < 4; ++t) {
      const _Float16* wr = Wt + (size_t)(t * 16 + m) * KPAD + k0 + half * 8;
      b[t].h[0] = *(const v8h*)(wr);
      b[t].h[1] = *(const v8h*)(wr + 16);
    }
#pragma unroll
    for (int t = 0; t < 4; ++t) {
      // (neg_a, A, neg_b, B, c_mod, C, reuse_a, reuse_b)
      c[t] = __builtin_amdgcn_wmma_f32_16x16x32_f16(false, a.v, false, b[t].v,
                                                    (short)0, c[t], false, false);
    }
  }

#pragma unroll
  for (int r = 0; r < 8; ++r) {
    const int orow = row0 + r + 8 * half;
    if (orow < N) {
      float* yr = Y + (size_t)orow * 64;
#pragma unroll
      for (int t = 0; t < 4; ++t) yr[t * 16 + m] = c[t][r];
    }
  }
}

// y[i] = dot(X[i,0:K], v)   (K in {16,32,64}, multiple of 4)
__global__ void rowdot(const float* __restrict__ X, const float* __restrict__ v,
                       float* __restrict__ y, int N, int K) {
  int i = blockIdx.x * blockDim.x + threadIdx.x;
  if (i >= N) return;
  const float* xr = X + (size_t)i * K;
  float s = 0.f;
  for (int j = 0; j < K; j += 4) {
    float4 xv = *(const float4*)(xr + j);
    s += xv.x * v[j] + xv.y * v[j + 1] + xv.z * v[j + 2] + xv.w * v[j + 3];
  }
  y[i] = s;
}

// out[j] = dot(Wd[j,0:64], ad)   (j < dd)  -- precompute Wd @ a_d
__global__ void colvec(const float* __restrict__ Wd, const float* __restrict__ ad,
                       float* __restrict__ out, int dd) {
  int j = blockIdx.x * blockDim.x + threadIdx.x;
  if (j >= dd) return;
  float s = 0.f;
  for (int h = 0; h < 64; ++h) s += Wd[(size_t)j * 64 + h] * ad[h];
  out[j] = s;
}

__global__ void fill_f32(float* p, float v, int n) {
  int i = blockIdx.x * blockDim.x + threadIdx.x;
  if (i < n) p[i] = v;
}
__global__ void fill_u32(unsigned* p, unsigned v, int n) {
  int i = blockIdx.x * blockDim.x + threadIdx.x;
  if (i < n) p[i] = v;
}

// monotonic uint encoding of float for atomicMax
__device__ __forceinline__ unsigned ordf(float f) {
  unsigned u = __float_as_uint(f);
  return (u >> 31) ? ~u : (u | 0x80000000u);
}
__device__ __forceinline__ float unordf(unsigned o) {
  return __uint_as_float((o >> 31) ? (o ^ 0x80000000u) : ~o);
}
#define ORD_NEG_INF 0x007FFFFFu  // ordf(-inf)

// e[k] = leaky_relu(ss[src]+sd[dst], 0.2); m[dst] = max(m[dst], e[k])
__global__ void edge_scores(const int* __restrict__ src, const int* __restrict__ dst,
                            const float* __restrict__ ss, const float* __restrict__ sd,
                            float* __restrict__ e, unsigned* __restrict__ m, int E) {
  int k = blockIdx.x * blockDim.x + threadIdx.x;
  if (k >= E) return;
  float x = ss[src[k]] + sd[dst[k]];
  float le = (x > 0.f) ? x : 0.2f * x;
  e[k] = le;
  atomicMax(&m[dst[k]], ordf(le));
}

// ex = exp(e - m[dst]); den[dst] += ex; acc[dst,h] += ex * hs[src,h]
// 16 threads per edge, 4 features each (b128 gather + 4 atomic f32 adds).
__global__ void edge_scatter(const int* __restrict__ src, const int* __restrict__ dst,
                             const float* __restrict__ e, const unsigned* __restrict__ m,
                             const float* __restrict__ hs, float* __restrict__ acc,
                             float* __restrict__ den, int E) {
  int t = blockIdx.x * blockDim.x + threadIdx.x;
  int k = t >> 4;
  int q = (t & 15) * 4;
  if (k >= E) return;
  int d = dst[k], s = src[k];
  float ex = __expf(e[k] - unordf(m[d]));
  if (q == 0) atomicAdd(&den[d], ex);
  float4 hv = *(const float4*)(hs + (size_t)s * 64 + q);
  float* ap = acc + (size_t)d * 64 + q;
  atomicAdd(ap + 0, ex * hv.x);
  atomicAdd(ap + 1, ex * hv.y);
  atomicAdd(ap + 2, ex * hv.z);
  atomicAdd(ap + 3, ex * hv.w);
}

// out[n,h] (+)= acc[n,h] / (den[n] + 1e-16) + b[h]
__global__ void finalize(const float* __restrict__ acc, const float* __restrict__ den,
                         const float* __restrict__ b, float* __restrict__ out,
                         int N, int accumulate) {
  int t = blockIdx.x * blockDim.x + threadIdx.x;
  if (t >= N * 64) return;
  int n = t >> 6, h = t & 63;
  float v = acc[t] / (den[n] + 1e-16f) + b[h];
  if (accumulate) out[t] += v; else out[t] = v;
}

// out[n,0:2] = t2[n,:] @ Wfc + bfc
__global__ void fc_head(const float* __restrict__ t2, const float* __restrict__ Wfc,
                        const float* __restrict__ bfc, float* __restrict__ out, int N) {
  int n = blockIdx.x * blockDim.x + threadIdx.x;
  if (n >= N) return;
  float s0 = bfc[0], s1 = bfc[1];
  const float* r = t2 + (size_t)n * 64;
  for (int h = 0; h < 64; ++h) {
    float v = r[h];
    s0 += v * Wfc[2 * h + 0];
    s1 += v * Wfc[2 * h + 1];
  }
  out[2 * n + 0] = s0;
  out[2 * n + 1] = s1;
}

static inline int cdiv(int a, int b) { return (a + b - 1) / b; }

extern "C" void kernel_launch(void* const* d_in, const int* in_sizes, int n_in,
                              void* d_out, int out_size, void* d_ws, size_t ws_size,
                              hipStream_t stream) {
  (void)in_sizes; (void)n_in; (void)out_size; (void)ws_size;

  const float* x_wallet = (const float*)d_in[0];
  const float* x_token  = (const float*)d_in[1];
  const float* x_dev    = (const float*)d_in[2];
  // relation param bases: each has _Ws,_Wd,_as,_ad,_b
  // g1ww=3, g1bu=8, g1se=13, g1cr=18, g2ww=23(dead: w2 unused), g2bu=28, g2se=33
  const float* Wfc = (const float*)d_in[38];
  const float* bfc = (const float*)d_in[39];
  const int* ww_src = (const int*)d_in[40];
  const int* ww_dst = (const int*)d_in[41];
  const int* bu_src = (const int*)d_in[42];
  const int* bu_dst = (const int*)d_in[43];
  const int* se_src = (const int*)d_in[44];
  const int* se_dst = (const int*)d_in[45];
  const int* cr_src = (const int*)d_in[46];
  const int* cr_dst = (const int*)d_in[47];

  // ---- workspace layout (float units) ----
  float* base = (float*)d_ws;
  size_t off = 0;
  float* hs   = base + off; off += (size_t)NW * 64;    // projected src features (per relation)
  float* w1   = base + off; off += (size_t)NW * 64;    // conv1 wallet output
  float* t1   = base + off; off += (size_t)NT * 64;    // conv1 token output
  float* t2   = base + off; off += (size_t)NT * 64;    // conv2 token output
  float* acc  = base + off; off += (size_t)NT * 64;    // per-relation numerator
  float* ss   = base + off; off += NW;                 // src attention score
  float* sd   = base + off; off += NW;                 // dst attention score
  unsigned* mbuf = (unsigned*)(base + off); off += NW; // segment max (ordered uint)
  float* den  = base + off; off += NW;                 // softmax denominator
  float* ebuf = base + off; off += E_WW;               // per-edge logits
  float* wdv  = base + off; off += 64;                 // Wd @ a_d
  _Float16* Xh = (_Float16*)(base + off); off += (size_t)NWPAD * 64 / 2; // packed f16 src
  _Float16* Wth = (_Float16*)(base + off); off += 64 * 64 / 2 + 16;      // packed f16 W^T

  // Run one GATConv given a pre-packed f16 source feature buffer Xh[Npad,Kpad].
  auto run_gat = [&](int Ns, int Ks, int Kpad, int Npad,
                     const float* xd, int Nd, int Kd,
                     const int* es, const int* ed, int E,
                     int pidx, float* accbuf, float* outbuf, int accum) {
    const float* Ws  = (const float*)d_in[pidx + 0];
    const float* Wd  = (const float*)d_in[pidx + 1];
    const float* as_ = (const float*)d_in[pidx + 2];
    const float* ad_ = (const float*)d_in[pidx + 3];
    const float* b_  = (const float*)d_in[pidx + 4];

    cvt_wt<<<cdiv(64 * Kpad, 256), 256, 0, stream>>>(Ws, Wth, Ks, Kpad);
    if (Kpad == 32)
      gemm_h64<32><<<Npad / 128, 256, 0, stream>>>(Xh, Wth, hs, Ns);
    else
      gemm_h64<64><<<Npad / 128, 256, 0, stream>>>(Xh, Wth, hs, Ns);
    rowdot     <<<cdiv(Ns, 256), 256, 0, stream>>>(hs, as_, ss, Ns, 64);
    colvec     <<<1, 64, 0, stream>>>(Wd, ad_, wdv, Kd);
    rowdot     <<<cdiv(Nd, 256), 256, 0, stream>>>(xd, wdv, sd, Nd, Kd);
    fill_u32   <<<cdiv(Nd, 256), 256, 0, stream>>>(mbuf, ORD_NEG_INF, Nd);
    fill_f32   <<<cdiv(Nd, 256), 256, 0, stream>>>(den, 0.f, Nd);
    fill_f32   <<<cdiv(Nd * 64, 256), 256, 0, stream>>>(accbuf, 0.f, Nd * 64);
    edge_scores<<<cdiv(E, 256), 256, 0, stream>>>(es, ed, ss, sd, ebuf, mbuf, E);
    edge_scatter<<<cdiv(E * 16, 256), 256, 0, stream>>>(es, ed, ebuf, mbuf, hs, accbuf, den, E);
    finalize   <<<cdiv(Nd * 64, 256), 256, 0, stream>>>(accbuf, den, b_, outbuf, Nd, accum);
  };

  // ---- conv1 ----
  // pack x_wallet once (Kpad=32); reused by ww, bu, se
  cvt_rows<<<cdiv(NWPAD * 32, 256), 256, 0, stream>>>(x_wallet, Xh, NW, 32, 32, NWPAD);
  run_gat(NW, 32, 32, NWPAD, x_wallet, NW, 32, ww_src, ww_dst, E_WW, 3, w1, w1, 0);
  fill_f32<<<cdiv(NT * 64, 256), 256, 0, stream>>>(t1, 0.f, NT * 64);
  run_gat(NW, 32, 32, NWPAD, x_token, NT, 32, bu_src, bu_dst, E_BU, 8,  acc, t1, 1);
  run_gat(NW, 32, 32, NWPAD, x_token, NT, 32, se_src, se_dst, E_SE, 13, acc, t1, 1);
  // pack x_dev (K=16 -> Kpad=32)
  cvt_rows<<<cdiv(NDPAD * 32, 256), 256, 0, stream>>>(x_dev, Xh, ND, 16, 32, NDPAD);
  run_gat(ND, 16, 32, NDPAD, x_token, NT, 32, cr_src, cr_dst, E_CR, 18, acc, t1, 1);

  // ---- conv2 ---- (w2 = gat(w1,w1,g2ww) is dead in the reference -> skipped)
  // pack w1 once (Kpad=64); reused by bu, se
  cvt_rows<<<cdiv(NWPAD * 64, 256), 256, 0, stream>>>(w1, Xh, NW, 64, 64, NWPAD);
  fill_f32<<<cdiv(NT * 64, 256), 256, 0, stream>>>(t2, 0.f, NT * 64);
  run_gat(NW, 64, 64, NWPAD, t1, NT, 64, bu_src, bu_dst, E_BU, 28, acc, t2, 1);
  run_gat(NW, 64, 64, NWPAD, t1, NT, 64, se_src, se_dst, E_SE, 33, acc, t2, 1);

  // ---- head ----
  fc_head<<<cdiv(NT, 256), 256, 0, stream>>>(t2, Wfc, bfc, (float*)d_out, NT);
}
